// CvT_54451595378946
// MI455X (gfx1250) — compile-verified, matrix operand order checked by reading, not run on hardware
//
#include <hip/hip_runtime.h>
#include <hip/hip_bf16.h>
#include <math.h>
#include <stdint.h>

// ---------- types ----------
typedef __bf16 bf16;
typedef __attribute__((ext_vector_type(16))) __bf16 v16bf;
typedef __attribute__((ext_vector_type(8)))  float  v8f;

__device__ __forceinline__ bf16 f2bf(float f) {
  unsigned u = __builtin_bit_cast(unsigned, f);
  unsigned r = (u + 0x7fffu + ((u >> 16) & 1u)) >> 16;
  unsigned short s = (unsigned short)r;
  return __builtin_bit_cast(bf16, s);
}
__device__ __forceinline__ float bf2f(bf16 b) {
  unsigned short s = __builtin_bit_cast(unsigned short, b);
  return __builtin_bit_cast(float, (unsigned)s << 16);
}

// async global->LDS copy of 16 bytes (per active lane), tracked by ASYNCcnt
__device__ __forceinline__ void async_b128(void* lds_dst, const void* gsrc) {
  unsigned lds = (unsigned)(uintptr_t)lds_dst;   // low 32 bits = LDS offset
  asm volatile("global_load_async_to_lds_b128 %0, %1, off"
               :: "v"(lds), "v"(gsrc) : "memory");
}
__device__ __forceinline__ void wait_async0() {
  asm volatile("s_wait_asynccnt 0x0" ::: "memory");
}
__device__ __forceinline__ void wait_async2() {
  asm volatile("s_wait_asynccnt 0x2" ::: "memory");
}

// ---------- problem constants ----------
constexpr int B_    = 32;
constexpr int C_    = 384;
constexpr int H_    = 28;
constexpr int W_    = 28;
constexpr int HEADS = 12;
constexpr int HD    = 32;      // head dim
constexpr int WIN   = 7;
constexpr int NWIN  = 49;      // tokens per window
constexpr int NW    = 16;      // windows per image (4x4)
constexpr int MPIX  = B_ * H_ * W_;            // 25088
constexpr float SCALE = 0.051031036307982884f; // 384^-0.5
constexpr float EPS = 1e-5f;

// ---------- layout converters ----------
__global__ __launch_bounds__(256) void nchw_to_nhwc(const float* __restrict__ src,
                                                    float* __restrict__ dst) {
  int idx = blockIdx.x * 256 + threadIdx.x;           // NHWC linear
  if (idx >= MPIX * C_) return;
  int c = idx % C_, pix = idx / C_;
  int b = pix / (H_ * W_), p = pix % (H_ * W_);
  dst[idx] = src[((size_t)b * C_ + c) * (H_ * W_) + p];
}

__global__ __launch_bounds__(256) void nhwc_to_nchw(const float* __restrict__ src,
                                                    float* __restrict__ dst) {
  int idx = blockIdx.x * 256 + threadIdx.x;           // NCHW linear
  if (idx >= MPIX * C_) return;
  int p = idx % (H_ * W_);
  int t = idx / (H_ * W_);
  int c = t % C_, b = t / C_;
  dst[idx] = src[((size_t)b * (H_ * W_) + p) * C_ + c];
}

__global__ __launch_bounds__(256) void f32_to_bf16_k(const float* __restrict__ in,
                                                     bf16* __restrict__ out, int n) {
  int i = blockIdx.x * 256 + threadIdx.x;
  if (i < n) out[i] = f2bf(in[i]);
}

// ---------- LayerNorm over channels (NHWC), one wave32 per pixel ----------
__global__ __launch_bounds__(256) void layernorm_bf16(const float* __restrict__ x,
                                                      const float* __restrict__ g,
                                                      const float* __restrict__ b,
                                                      bf16* __restrict__ out) {
  int wave = threadIdx.x >> 5, lane = threadIdx.x & 31;
  int pix = blockIdx.x * 8 + wave;
  if (pix >= MPIX) return;
  const float* row = x + (size_t)pix * C_;
  float s = 0.f, s2 = 0.f;
  for (int c = lane; c < C_; c += 32) { float v = row[c]; s += v; s2 += v * v; }
  for (int o = 16; o; o >>= 1) { s += __shfl_xor(s, o, 32); s2 += __shfl_xor(s2, o, 32); }
  float mean = s * (1.f / C_);
  float var = s2 * (1.f / C_) - mean * mean;
  float inv = rsqrtf(var + EPS);
  bf16* orow = out + (size_t)pix * C_;
  for (int c = lane; c < C_; c += 32)
    orow[c] = f2bf((row[c] - mean) * inv * g[c] + b[c]);
}

// ---------- depthwise 3x3 conv + BatchNorm(eval) ----------
__global__ __launch_bounds__(256) void dwconv_bn(const bf16* __restrict__ xn,
                                                 const float* __restrict__ dww,
                                                 const float* __restrict__ bng,
                                                 const float* __restrict__ bnb,
                                                 const float* __restrict__ bnm,
                                                 const float* __restrict__ bnv,
                                                 bf16* __restrict__ out) {
  int idx = blockIdx.x * 256 + threadIdx.x;
  if (idx >= MPIX * C_) return;
  int c = idx % C_;
  int pix = idx / C_;
  int w = pix % W_, t = pix / W_;
  int h = t % H_, b = t / H_;
  float acc = 0.f;
#pragma unroll
  for (int dy = -1; dy <= 1; ++dy)
#pragma unroll
    for (int dx = -1; dx <= 1; ++dx) {
      int hh = h + dy, ww = w + dx;
      if (hh < 0 || hh >= H_ || ww < 0 || ww >= W_) continue;
      float wv = dww[c * 9 + (dy + 1) * 3 + (dx + 1)];
      acc += wv * bf2f(xn[((((size_t)b * H_ + hh) * W_) + ww) * C_ + c]);
    }
  float r = (acc - bnm[c]) * rsqrtf(bnv[c] + EPS) * bng[c] + bnb[c];
  out[idx] = f2bf(r);
}

// ---------- WMMA GEMM: out[M,N] = A[M,K] * W[N,K]^T + bias, fused epilogues ----------
// Double-buffered async(global->LDS) pipeline; 128x128 block tile; 8 wave32.
// MODE 0: QKV scatter into window layout (q scaled by SCALE)
// MODE 1: + residual -> f32 (proj)
// MODE 2: GELU(exact) -> bf16 (FFN1)
// MODE 3: + residual -> f32 (FFN2)
template <int MODE>
__global__ __launch_bounds__(256) void gemm_wmma(
    const bf16* __restrict__ A, const bf16* __restrict__ W,
    const float* __restrict__ bias, int M, int N, int K,
    const float* __restrict__ res, float* __restrict__ outf,
    bf16* __restrict__ outb,
    bf16* __restrict__ qout, bf16* __restrict__ kout, bf16* __restrict__ vout) {
  __shared__ __align__(16) bf16 As[2][128][40];   // 80B row stride: b128-aligned, conflict-free
  __shared__ __align__(16) bf16 Bs[2][128][40];
  const int tid = threadIdx.x;
  const int bm = blockIdx.y, bn = blockIdx.x;
  const int wave = tid >> 5, lane = tid & 31;
  const int wm = wave & 3, wn = wave >> 2;             // 4x2 wave grid over 128x128
  const int fm = lane & 15;
  const int kbase = (lane >> 4) * 8;                   // A/B fragment K base per ISA layout
  const int rofs = (lane >> 4) * 8;                    // C/D row offset per ISA layout
  const int lrow = tid >> 1, lseg = tid & 1;

  v8f acc[2][4];
#pragma unroll
  for (int i = 0; i < 2; ++i)
#pragma unroll
    for (int j = 0; j < 4; ++j)
#pragma unroll
      for (int e = 0; e < 8; ++e) acc[i][j][e] = 0.f;

  const bf16* aptr = A + (size_t)(bm * 128 + lrow) * K + lseg * 16;
  const bf16* wptr = W + (size_t)(bn * 128 + lrow) * K + lseg * 16;

  auto issue = [&](int kk, int buf) {
    async_b128(&As[buf][lrow][lseg * 16], aptr + kk);
    async_b128(&Bs[buf][lrow][lseg * 16], wptr + kk);
  };

  const int nk = K >> 5;
  issue(0, 0);
  for (int s = 0; s < nk; ++s) {
    const int cur = s & 1;
    if (s + 1 < nk) {
      issue((s + 1) * 32, cur ^ 1);   // prefetch next slab via async DMA
      wait_async2();                  // wait for current slab only (in-order completion)
    } else {
      wait_async0();
    }
    __syncthreads();

    v16bf af[2], bfv[4];
#pragma unroll
    for (int i = 0; i < 2; ++i) {
      int r = wm * 32 + i * 16 + fm;
#pragma unroll
      for (int e = 0; e < 8; ++e) {
        af[i][e] = As[cur][r][kbase + e];
        af[i][8 + e] = As[cur][r][kbase + 16 + e];
      }
    }
#pragma unroll
    for (int j = 0; j < 4; ++j) {
      int c = wn * 64 + j * 16 + fm;
#pragma unroll
      for (int e = 0; e < 8; ++e) {
        bfv[j][e] = Bs[cur][c][kbase + e];
        bfv[j][8 + e] = Bs[cur][c][kbase + 16 + e];
      }
    }
#pragma unroll
    for (int i = 0; i < 2; ++i)
#pragma unroll
      for (int j = 0; j < 4; ++j)
        acc[i][j] = __builtin_amdgcn_wmma_f32_16x16x32_bf16(
            false, af[i], false, bfv[j], (short)0, acc[i][j], false, false);
    __syncthreads();
  }

#pragma unroll
  for (int i = 0; i < 2; ++i)
#pragma unroll
    for (int j = 0; j < 4; ++j) {
      int col = bn * 128 + wn * 64 + j * 16 + fm;
      float bv = bias[col];
#pragma unroll
      for (int r = 0; r < 8; ++r) {
        int row = bm * 128 + wm * 32 + i * 16 + rofs + r;
        float v = acc[i][j][r] + bv;
        if constexpr (MODE == 0) {
          int which = col / C_, c = col % C_;
          int head = c >> 5, d = c & 31;
          int bimg = row / (H_ * W_), pix = row % (H_ * W_);
          int hh = pix / W_, ww = pix % W_;
          int sxw = hh / WIN, wx = hh % WIN, syw = ww / WIN, wy = ww % WIN;
          int wiw = bimg * NW + sxw * 4 + syw;
          size_t dst = (((size_t)wiw * HEADS + head) * NWIN + (wx * WIN + wy)) * HD + d;
          if (which == 0)      qout[dst] = f2bf(v * SCALE);
          else if (which == 1) kout[dst] = f2bf(v);
          else                 vout[dst] = f2bf(v);
        } else if constexpr (MODE == 1 || MODE == 3) {
          size_t o = (size_t)row * C_ + col;
          outf[o] = v + res[o];
        } else {
          float g = 0.5f * v * (1.f + erff(v * 0.70710678118654752f));
          outb[(size_t)row * N + col] = f2bf(g);
        }
      }
    }
}

// ---------- windowed attention: one block per (window, head), 4 waves ----------
__global__ __launch_bounds__(128) void attn_wmma(const bf16* __restrict__ qb,
                                                 const bf16* __restrict__ kb,
                                                 const bf16* __restrict__ vb,
                                                 const float* __restrict__ rpb,
                                                 bf16* __restrict__ outb) {
  __shared__ __align__(16) bf16 qs[64][40];
  __shared__ __align__(16) bf16 ks[64][40];
  __shared__ __align__(16) bf16 vs[64][40];
  __shared__ float dots[64][68];
  __shared__ bf16 att[64][72];

  const int tid = threadIdx.x;
  const int blk = blockIdx.x;                   // wi*HEADS + head
  const int wi = blk / HEADS, head = blk % HEADS;
  const int b = wi >> 4, win = wi & 15, sx = win >> 2, sy = win & 3;

  const bf16* qsrc = qb + (size_t)blk * NWIN * HD;
  const bf16* ksrc = kb + (size_t)blk * NWIN * HD;
  const bf16* vsrc = vb + (size_t)blk * NWIN * HD;

  // 1) kick off async DMA of q/k/v valid rows (49 rows x 64B = 4 x b128 each)
  for (int s = tid; s < 3 * NWIN * 4; s += 128) {
    int t = s / (NWIN * 4), r = s % (NWIN * 4);
    int row = r >> 2, seg = r & 3;
    const bf16* src = (t == 0 ? qsrc : (t == 1 ? ksrc : vsrc)) + row * HD + seg * 8;
    bf16* dst = (t == 0 ? &qs[row][seg * 8] : (t == 1 ? &ks[row][seg * 8] : &vs[row][seg * 8]));
    async_b128(dst, src);
  }
  // 2) zero pad rows 49..63 (overlaps with DMA)
  bf16 z = f2bf(0.f);
  for (int idx = tid; idx < (64 - NWIN) * 32; idx += 128) {
    int n = NWIN + (idx >> 5), d = idx & 31;
    qs[n][d] = z; ks[n][d] = z; vs[n][d] = z;
  }
  // 3) precompute (rel-pos bias + shift mask + col-pad) table into dots (overlaps DMA)
  for (int idx = tid; idx < 64 * 64; idx += 128) {
    int row = idx >> 6, col = idx & 63;
    float v;
    if (col >= NWIN) {
      v = -1e9f;                                 // pad cols -> exp() == 0
    } else if (row >= NWIN) {
      v = 0.f;
    } else {
      int ridx = (row / WIN - col / WIN + 6) * 13 + (row % WIN - col % WIN + 6);
      v = rpb[ridx * HEADS + head];              // relative position bias
      int hr = sx * WIN + row / WIN, wr = sy * WIN + row % WIN;
      int hc = sx * WIN + col / WIN, wc = sy * WIN + col % WIN;
      int idr = (hr < 21 ? 0 : (hr < 25 ? 1 : 2)) * 3 + (wr < 21 ? 0 : (wr < 25 ? 1 : 2));
      int idc = (hc < 21 ? 0 : (hc < 25 ? 1 : 2)) * 3 + (wc < 21 ? 0 : (wc < 25 ? 1 : 2));
      if (idr != idc) v -= 100.f;                // Swin shift mask
    }
    dots[row][col] = v;
  }
  wait_async0();
  __syncthreads();

  const int wave = tid >> 5, lane = tid & 31;
  const int fm = lane & 15, kbase = (lane >> 4) * 8, rofs = (lane >> 4) * 8;

  // ---- dots += q @ k^T, rows 16*wave..+15 ----
  v16bf aq;
  {
    int m = wave * 16 + fm;
#pragma unroll
    for (int e = 0; e < 8; ++e) { aq[e] = qs[m][kbase + e]; aq[8 + e] = qs[m][kbase + 16 + e]; }
  }
#pragma unroll
  for (int nt = 0; nt < 4; ++nt) {
    v16bf bk;
    int n = nt * 16 + fm;
#pragma unroll
    for (int e = 0; e < 8; ++e) { bk[e] = ks[n][kbase + e]; bk[8 + e] = ks[n][kbase + 16 + e]; }
    v8f acc;
#pragma unroll
    for (int e = 0; e < 8; ++e) acc[e] = 0.f;
    acc = __builtin_amdgcn_wmma_f32_16x16x32_bf16(false, aq, false, bk, (short)0, acc, false, false);
    int col = nt * 16 + fm;
#pragma unroll
    for (int r = 0; r < 8; ++r) {
      int row = wave * 16 + rofs + r;
      dots[row][col] += acc[r];                  // bias/mask already in table
    }
  }
  __syncthreads();

  // ---- softmax per row (f32) -> att (bf16) ----
  if (tid < 64) {
    int row = tid;
    float mx = -1e30f;
    for (int j = 0; j < 64; ++j) mx = fmaxf(mx, dots[row][j]);
    float sum = 0.f;
    for (int j = 0; j < 64; ++j) sum += __expf(dots[row][j] - mx);
    float inv = 1.f / sum;
    for (int j = 0; j < 64; ++j) att[row][j] = f2bf(__expf(dots[row][j] - mx) * inv);
  }
  __syncthreads();

  // ---- out = att @ v, unwindow scatter ----
#pragma unroll
  for (int nt = 0; nt < 2; ++nt) {
    v8f acc;
#pragma unroll
    for (int e = 0; e < 8; ++e) acc[e] = 0.f;
#pragma unroll
    for (int ks2 = 0; ks2 < 2; ++ks2) {
      v16bf aa, bv;
      int m = wave * 16 + fm;
      int kb0 = ks2 * 32 + kbase;
#pragma unroll
      for (int e = 0; e < 8; ++e) { aa[e] = att[m][kb0 + e]; aa[8 + e] = att[m][kb0 + 16 + e]; }
      int n = nt * 16 + fm;
#pragma unroll
      for (int e = 0; e < 8; ++e) { bv[e] = vs[kb0 + e][n]; bv[8 + e] = vs[kb0 + 16 + e][n]; }
      acc = __builtin_amdgcn_wmma_f32_16x16x32_bf16(false, aa, false, bv, (short)0, acc, false, false);
    }
    int col = nt * 16 + fm;                      // d within head
#pragma unroll
    for (int r = 0; r < 8; ++r) {
      int row = wave * 16 + rofs + r;            // token within window
      if (row < NWIN) {
        int wx = row / WIN, wy = row % WIN;
        int hh = sx * WIN + wx, ww = sy * WIN + wy;
        size_t Mp = ((size_t)b * H_ + hh) * W_ + ww;
        outb[Mp * C_ + head * HD + col] = f2bf(acc[r]);
      }
    }
  }
}

// ---------- host side ----------
extern "C" void kernel_launch(void* const* d_in, const int* in_sizes, int n_in,
                              void* d_out, int out_size, void* d_ws, size_t ws_size,
                              hipStream_t stream) {
  const float* x_in   = (const float*)d_in[0];
  const float* ln1_g  = (const float*)d_in[1];
  const float* ln1_b  = (const float*)d_in[2];
  const float* dw_w   = (const float*)d_in[3];
  const float* bn_g   = (const float*)d_in[4];
  const float* bn_b   = (const float*)d_in[5];
  const float* bn_m   = (const float*)d_in[6];
  const float* bn_v   = (const float*)d_in[7];
  const float* pw_w   = (const float*)d_in[8];
  const float* pw_b   = (const float*)d_in[9];
  const float* proj_w = (const float*)d_in[10];
  const float* proj_b = (const float*)d_in[11];
  const float* rpb    = (const float*)d_in[12];
  const float* ln2_g  = (const float*)d_in[13];
  const float* ln2_b  = (const float*)d_in[14];
  const float* ff_w1  = (const float*)d_in[15];
  const float* ff_b1  = (const float*)d_in[16];
  const float* ff_w2  = (const float*)d_in[17];
  const float* ff_b2  = (const float*)d_in[18];
  (void)in_sizes; (void)n_in; (void)out_size; (void)ws_size;

  char* ws = (char*)d_ws;
  size_t off = 0;
  auto take = [&](size_t bytes) {
    char* p = ws + off;
    off += (bytes + 255) & ~(size_t)255;
    return p;
  };
  float* x_nhwc = (float*)take((size_t)MPIX * C_ * 4);
  bf16* xn_bf   = (bf16*)take((size_t)MPIX * C_ * 2);
  bf16* t_bf    = (bf16*)take((size_t)MPIX * C_ * 2);
  bf16* qbuf    = (bf16*)take((size_t)B_ * NW * HEADS * NWIN * HD * 2);
  bf16* kbuf    = (bf16*)take((size_t)B_ * NW * HEADS * NWIN * HD * 2);
  bf16* vbuf    = (bf16*)take((size_t)B_ * NW * HEADS * NWIN * HD * 2);
  bf16* ao_bf   = (bf16*)take((size_t)MPIX * C_ * 2);
  bf16* h1_bf   = (bf16*)take((size_t)MPIX * 4 * C_ * 2);
  bf16* wpw     = (bf16*)take((size_t)3 * C_ * C_ * 2);
  bf16* wproj   = (bf16*)take((size_t)C_ * C_ * 2);
  bf16* wff1    = (bf16*)take((size_t)4 * C_ * C_ * 2);
  bf16* wff2    = (bf16*)take((size_t)C_ * 4 * C_ * 2);

  const int total = MPIX * C_;                 // 9,633,792
  const int g256 = (total + 255) / 256;        // 37632

  nchw_to_nhwc<<<dim3(g256), 256, 0, stream>>>(x_in, x_nhwc);

  for (int i = 0; i < 2; ++i) {
    // weights -> bf16
    f32_to_bf16_k<<<dim3((3 * C_ * C_ + 255) / 256), 256, 0, stream>>>(
        pw_w + (size_t)i * 3 * C_ * C_, wpw, 3 * C_ * C_);
    f32_to_bf16_k<<<dim3((C_ * C_ + 255) / 256), 256, 0, stream>>>(
        proj_w + (size_t)i * C_ * C_, wproj, C_ * C_);
    f32_to_bf16_k<<<dim3((4 * C_ * C_ + 255) / 256), 256, 0, stream>>>(
        ff_w1 + (size_t)i * 4 * C_ * C_, wff1, 4 * C_ * C_);
    f32_to_bf16_k<<<dim3((4 * C_ * C_ + 255) / 256), 256, 0, stream>>>(
        ff_w2 + (size_t)i * 4 * C_ * C_, wff2, 4 * C_ * C_);

    // LN1 -> dwconv+BN -> QKV GEMM -> attention -> proj(+res)
    layernorm_bf16<<<dim3(MPIX / 8), 256, 0, stream>>>(
        x_nhwc, ln1_g + i * C_, ln1_b + i * C_, xn_bf);
    dwconv_bn<<<dim3(g256), 256, 0, stream>>>(
        xn_bf, dw_w + (size_t)i * C_ * 9, bn_g + i * C_, bn_b + i * C_,
        bn_m + i * C_, bn_v + i * C_, t_bf);
    gemm_wmma<0><<<dim3((3 * C_) / 128, MPIX / 128), 256, 0, stream>>>(
        t_bf, wpw, pw_b + (size_t)i * 3 * C_, MPIX, 3 * C_, C_,
        nullptr, nullptr, nullptr, qbuf, kbuf, vbuf);
    attn_wmma<<<dim3(B_ * NW * HEADS), 128, 0, stream>>>(
        qbuf, kbuf, vbuf, rpb + (size_t)i * 169 * HEADS, ao_bf);
    gemm_wmma<1><<<dim3(C_ / 128, MPIX / 128), 256, 0, stream>>>(
        ao_bf, wproj, proj_b + (size_t)i * C_, MPIX, C_, C_,
        x_nhwc, x_nhwc, nullptr, nullptr, nullptr, nullptr);

    // LN2 -> FFN1(GELU) -> FFN2(+res)
    layernorm_bf16<<<dim3(MPIX / 8), 256, 0, stream>>>(
        x_nhwc, ln2_g + i * C_, ln2_b + i * C_, xn_bf);
    gemm_wmma<2><<<dim3((4 * C_) / 128, MPIX / 128), 256, 0, stream>>>(
        xn_bf, wff1, ff_b1 + (size_t)i * 4 * C_, MPIX, 4 * C_, C_,
        nullptr, nullptr, h1_bf, nullptr, nullptr, nullptr);
    gemm_wmma<3><<<dim3(C_ / 128, MPIX / 128), 256, 0, stream>>>(
        h1_bf, wff2, ff_b2 + (size_t)i * C_, MPIX, C_, 4 * C_,
        x_nhwc, x_nhwc, nullptr, nullptr, nullptr, nullptr);
  }

  nhwc_to_nchw<<<dim3(g256), 256, 0, stream>>>(x_nhwc, (float*)d_out);
}
